// Token_QK_Attention_43344809952057
// MI455X (gfx1250) — matrix operand order repaired and checked by reading, not compiled, not run on hardware
//
#include <hip/hip_runtime.h>
#include <hip/hip_bf16.h>

// ---------------------------------------------------------------------------
// Token QK attention (spiking) for gfx1250.
// Sizes fixed by the reference: T=4, B=16, C=512, N=1024, HEADS=8, Dh=64.
// Pipeline:
//   1) prep:       fp32 weights -> f16, fold BN (and proj bias) into scale/bias
//   2) lif1:       LIF over T on x, LDS transpose -> xsT f16 [T,B,N,C]
//   3) qk gemm:    WMMA f16 GEMMs for q & k; B tiles staged via double-buffered
//                  LDS (shared by all 8 waves), all 4 t in regs, BN+LIF epilogue
//   4) attn:       head-sum, LIF(vth=0.5) -> attn out; yT = attn * k
//   5) proj gemm:  WMMA f16 GEMM (LDS-staged B), scale/bias -> fp32 out
// ---------------------------------------------------------------------------

#define TT 4
#define BB 16
#define CC 512
#define NN 1024
#define HEADS 8
#define DH 64
#define LDSPITCH 40   // halves per LDS row (32 data + 8 pad, 80B = 16B aligned)

typedef __attribute__((ext_vector_type(16))) _Float16 v16h;
typedef __attribute__((ext_vector_type(8)))  _Float16 v8h;
typedef __attribute__((ext_vector_type(4)))  _Float16 v4h;
typedef __attribute__((ext_vector_type(8)))  float    v8f;

union U16h { v16h v; v8h h[2]; };

// ---------------------------------------------------------------- prep ------
__global__ void __launch_bounds__(256)
prep_kernel(const float* __restrict__ qw, const float* __restrict__ kw,
            const float* __restrict__ pw,
            const float* __restrict__ qg, const float* __restrict__ qbeta,
            const float* __restrict__ qm, const float* __restrict__ qv,
            const float* __restrict__ kg, const float* __restrict__ kbeta,
            const float* __restrict__ km, const float* __restrict__ kv,
            const float* __restrict__ projb,
            const float* __restrict__ pg, const float* __restrict__ pbeta,
            const float* __restrict__ pm, const float* __restrict__ pv,
            _Float16* __restrict__ wqh, _Float16* __restrict__ wkh,
            _Float16* __restrict__ wph,
            float* __restrict__ qscale, float* __restrict__ qbias,
            float* __restrict__ kscale, float* __restrict__ kbias,
            float* __restrict__ pscale, float* __restrict__ pbias)
{
    int i = blockIdx.x * blockDim.x + threadIdx.x;
    if (i < CC * CC) {
        wqh[i] = (_Float16)qw[i];
        wkh[i] = (_Float16)kw[i];
        wph[i] = (_Float16)pw[i];
    }
    if (i < CC) {
        float inv = qg[i] * rsqrtf(qv[i] + 1e-5f);
        qscale[i] = inv; qbias[i] = qbeta[i] - qm[i] * inv;
        inv = kg[i] * rsqrtf(kv[i] + 1e-5f);
        kscale[i] = inv; kbias[i] = kbeta[i] - km[i] * inv;
        inv = pg[i] * rsqrtf(pv[i] + 1e-5f);
        pscale[i] = inv; pbias[i] = pbeta[i] - pm[i] * inv + projb[i] * inv;
    }
}

// --------------------------------------------------- LIF1 + transpose ------
// x [T,B,C,N] fp32 -> xsT [T,B,N,C] f16 spikes. LIF: v=(v+x)/2; s=(v>=1); v*=(1-s)
__global__ void __launch_bounds__(256)
lif_transpose_kernel(const float* __restrict__ x, _Float16* __restrict__ xsT)
{
    __shared__ _Float16 tile[32][33];
    const int nb = blockIdx.x * 32, cb = blockIdx.y * 32, b = blockIdx.z;
    const int tx = threadIdx.x, ty = threadIdx.y;   // block (32,8)
    float v[4] = {0.f, 0.f, 0.f, 0.f};
    for (int t = 0; t < TT; ++t) {
        #pragma unroll
        for (int i = 0; i < 4; ++i) {
            int c = cb + ty + i * 8;
            float xv = x[(((size_t)t * BB + b) * CC + c) * NN + nb + tx];
            v[i] = (v[i] + xv) * 0.5f;
            float s = (v[i] >= 1.0f) ? 1.0f : 0.0f;
            v[i] -= v[i] * s;
            tile[ty + i * 8][tx] = (_Float16)s;      // tile[c_local][n_local]
        }
        __syncthreads();
        #pragma unroll
        for (int i = 0; i < 4; ++i) {
            int nl = ty + i * 8;
            xsT[(((size_t)t * BB + b) * NN + nb + nl) * CC + cb + tx] = tile[tx][nl];
        }
        __syncthreads();
    }
}

// ------------------------------------------- fused Q/K GEMM + BN + LIF -----
// D[d,n] = sum_c W[d,c]*xsT[n,c] per (t,b); all 4 t in registers, LIF epilogue.
// grid (NN/32, CC/128, BB), block 256 (8 waves); wave -> 16 d rows x 32 n cols.
// B tiles are staged once per block through double-buffered LDS.
__global__ void __launch_bounds__(256)
qk_gemm_lif_kernel(const _Float16* __restrict__ xsT,
                   const _Float16* __restrict__ wq,
                   const _Float16* __restrict__ wk,
                   const float* __restrict__ qscale, const float* __restrict__ qbias,
                   const float* __restrict__ kscale, const float* __restrict__ kbias,
                   _Float16* __restrict__ qT, _Float16* __restrict__ kT)
{
    __shared__ _Float16 bstage[2][TT][32][LDSPITCH];   // 2 x 4t x 32n x 32k (+pad)

    const int tid  = threadIdx.x;
    const int lane = tid & 31;
    const int wave = tid >> 5;
    const int b     = blockIdx.z;
    const int nbase = blockIdx.x * 32;
    const int dwave = blockIdx.y * 128 + wave * 16;
    const int hi = lane >> 4;          // half-wave group
    const int ml = lane & 15;

    // staging role: thread -> (t, n_local, k_half) covering [4][32][32] halves
    const int st_t  = tid >> 6;
    const int st_nl = (tid >> 1) & 31;
    const int st_kh = (tid & 1) * 16;
    const _Float16* st_src =
        xsT + (((size_t)st_t * BB + b) * NN + nbase + st_nl) * CC + st_kh;

    v8f zacc = {};
    v8f accq[TT][2], acck[TT][2];
    #pragma unroll
    for (int t = 0; t < TT; ++t) { accq[t][0]=zacc; accq[t][1]=zacc; acck[t][0]=zacc; acck[t][1]=zacc; }

    // A-fragment base: lane hi=0 holds K {0..7,16..23}, hi=1 holds K {8..15,24..31}
    const _Float16* aq_base = wq + (size_t)(dwave + ml) * CC + hi * 8;
    const _Float16* ak_base = wk + (size_t)(dwave + ml) * CC + hi * 8;

    for (int kc = 0; kc < CC; kc += 32) {
        const int buf = (kc >> 5) & 1;
        // ---- stage B tile (all 4 t) into LDS: 32B per thread ----
        {
            v8h s0 = *(const v8h*)(st_src + kc);
            v8h s1 = *(const v8h*)(st_src + kc + 8);
            if (kc + 32 < CC) __builtin_prefetch(st_src + kc + 32, 0, 1);
            *(v8h*)&bstage[buf][st_t][st_nl][st_kh]     = s0;
            *(v8h*)&bstage[buf][st_t][st_nl][st_kh + 8] = s1;
        }
        // ---- A fragments from global (weights, L2 resident) ----
        U16h aq, ak;
        aq.h[0] = *(const v8h*)(aq_base + kc);
        aq.h[1] = *(const v8h*)(aq_base + kc + 16);
        ak.h[0] = *(const v8h*)(ak_base + kc);
        ak.h[1] = *(const v8h*)(ak_base + kc + 16);
        __syncthreads();
        #pragma unroll
        for (int t = 0; t < TT; ++t) {
            #pragma unroll
            for (int s = 0; s < 2; ++s) {
                U16h bf;
                bf.h[0] = *(const v8h*)&bstage[buf][t][s * 16 + ml][hi * 8];
                bf.h[1] = *(const v8h*)&bstage[buf][t][s * 16 + ml][hi * 8 + 16];
                accq[t][s] = __builtin_amdgcn_wmma_f32_16x16x32_f16(
                    false, aq.v, false, bf.v, (short)0, accq[t][s], false, false);
                acck[t][s] = __builtin_amdgcn_wmma_f32_16x16x32_f16(
                    false, ak.v, false, bf.v, (short)0, acck[t][s], false, false);
            }
        }
    }

    // Epilogue: BN then sequential LIF over t, store transposed f16 spikes.
    const int dlane = dwave + hi * 8;   // + r -> channel d for accumulator VGPR r
    float sq[8], bq[8], sk[8], bk[8];
    #pragma unroll
    for (int r = 0; r < 8; ++r) {
        sq[r] = qscale[dlane + r]; bq[r] = qbias[dlane + r];
        sk[r] = kscale[dlane + r]; bk[r] = kbias[dlane + r];
    }
    #pragma unroll
    for (int s = 0; s < 2; ++s) {
        v8h outq[TT], outk[TT];
        #pragma unroll
        for (int r = 0; r < 8; ++r) {
            float vq = 0.f, vk = 0.f;
            #pragma unroll
            for (int t = 0; t < TT; ++t) {
                float uq = accq[t][s][r] * sq[r] + bq[r];
                vq = (vq + uq) * 0.5f;
                float sp = (vq >= 1.0f) ? 1.0f : 0.0f;
                vq -= vq * sp;
                outq[t][r] = (_Float16)sp;
                float uk = acck[t][s][r] * sk[r] + bk[r];
                vk = (vk + uk) * 0.5f;
                sp = (vk >= 1.0f) ? 1.0f : 0.0f;
                vk -= vk * sp;
                outk[t][r] = (_Float16)sp;
            }
        }
        const int n = nbase + s * 16 + ml;
        #pragma unroll
        for (int t = 0; t < TT; ++t) {
            size_t off = (((size_t)t * BB + b) * NN + n) * CC + dlane;
            *(v8h*)(qT + off) = outq[t];
            *(v8h*)(kT + off) = outk[t];
        }
    }
}

// ------------------------------------------------ attn LIF + y = attn*k ----
// thread per (b,h,n); everything contiguous in the [T,B,N,C] layout.
__global__ void __launch_bounds__(256)
attn_y_kernel(const _Float16* __restrict__ qT, const _Float16* __restrict__ kT,
              _Float16* __restrict__ yT, float* __restrict__ attn_out)
{
    int idx = blockIdx.x * blockDim.x + threadIdx.x;
    if (idx >= BB * HEADS * NN) return;
    const int n = idx & (NN - 1);
    const int h = (idx >> 10) & (HEADS - 1);
    const int b = idx >> 13;
    float v = 0.f;
    for (int t = 0; t < TT; ++t) {
        size_t base = (((size_t)t * BB + b) * NN + n) * CC + h * DH;
        float qs = 0.f;
        #pragma unroll
        for (int j = 0; j < DH; j += 8) {
            v8h qv = *(const v8h*)(qT + base + j);
            #pragma unroll
            for (int e = 0; e < 8; ++e) qs += (float)qv[e];
        }
        v = (v + qs) * 0.5f;                 // LIF charge
        float s = (v >= 0.5f) ? 1.0f : 0.0f; // vth = 0.5
        v -= v * s;
        attn_out[(((size_t)t * BB + b) * HEADS + h) * NN + n] = s;
        _Float16 sh = (_Float16)s;
        #pragma unroll
        for (int j = 0; j < DH; j += 8) {
            v8h kv = *(const v8h*)(kT + base + j);
            v8h yv;
            #pragma unroll
            for (int e = 0; e < 8; ++e) yv[e] = (_Float16)(kv[e] * sh);
            *(v8h*)(yT + base + j) = yv;
        }
    }
}

// ----------------------------------------------- proj GEMM + bias + BN -----
// grid (NN/32, CC/128, TT*BB); fp32 output in [T,B,C,N] order.
// B tiles staged through double-buffered LDS (8B per thread per chunk).
__global__ void __launch_bounds__(256)
proj_gemm_kernel(const _Float16* __restrict__ yT, const _Float16* __restrict__ wp,
                 const float* __restrict__ pscale, const float* __restrict__ pbias,
                 float* __restrict__ out)
{
    __shared__ _Float16 bstage[2][32][LDSPITCH];   // 2 x 32n x 32k (+pad)

    const int tid  = threadIdx.x;
    const int lane = tid & 31;
    const int wave = tid >> 5;
    const int tb    = blockIdx.z;        // t*BB + b
    const int nbase = blockIdx.x * 32;
    const int dwave = blockIdx.y * 128 + wave * 16;
    const int hi = lane >> 4, ml = lane & 15;

    // staging role: thread -> (n_local, 4-half group) covering [32][32] halves
    const int st_nl = tid >> 3;
    const int st_kh = (tid & 7) * 4;
    const _Float16* st_src = yT + ((size_t)tb * NN + nbase + st_nl) * CC + st_kh;

    v8f zacc = {};
    v8f acc[2]; acc[0] = zacc; acc[1] = zacc;
    const _Float16* ap = wp + (size_t)(dwave + ml) * CC + hi * 8;

    for (int kc = 0; kc < CC; kc += 32) {
        const int buf = (kc >> 5) & 1;
        {
            v4h s0 = *(const v4h*)(st_src + kc);
            if (kc + 32 < CC) __builtin_prefetch(st_src + kc + 32, 0, 1);
            *(v4h*)&bstage[buf][st_nl][st_kh] = s0;
        }
        U16h a;
        a.h[0] = *(const v8h*)(ap + kc);
        a.h[1] = *(const v8h*)(ap + kc + 16);
        __syncthreads();
        #pragma unroll
        for (int s = 0; s < 2; ++s) {
            U16h bf;
            bf.h[0] = *(const v8h*)&bstage[buf][s * 16 + ml][hi * 8];
            bf.h[1] = *(const v8h*)&bstage[buf][s * 16 + ml][hi * 8 + 16];
            acc[s] = __builtin_amdgcn_wmma_f32_16x16x32_f16(
                false, a.v, false, bf.v, (short)0, acc[s], false, false);
        }
    }

    const int dlane = dwave + hi * 8;
    float sc[8], bi[8];
    #pragma unroll
    for (int r = 0; r < 8; ++r) { sc[r] = pscale[dlane + r]; bi[r] = pbias[dlane + r]; }
    #pragma unroll
    for (int s = 0; s < 2; ++s) {
        const int n = nbase + s * 16 + ml;
        #pragma unroll
        for (int r = 0; r < 8; ++r)
            out[((size_t)tb * CC + dlane + r) * NN + n] = acc[s][r] * sc[r] + bi[r];
    }
}

// --------------------------------------------------------------------------
extern "C" void kernel_launch(void* const* d_in, const int* in_sizes, int n_in,
                              void* d_out, int out_size, void* d_ws, size_t ws_size,
                              hipStream_t stream) {
    const float* x      = (const float*)d_in[0];
    const float* q_w    = (const float*)d_in[1];
    const float* q_g    = (const float*)d_in[2];
    const float* q_b    = (const float*)d_in[3];
    const float* q_m    = (const float*)d_in[4];
    const float* q_v    = (const float*)d_in[5];
    const float* k_w    = (const float*)d_in[6];
    const float* k_g    = (const float*)d_in[7];
    const float* k_b    = (const float*)d_in[8];
    const float* k_m    = (const float*)d_in[9];
    const float* k_v    = (const float*)d_in[10];
    const float* proj_w = (const float*)d_in[11];
    const float* proj_b = (const float*)d_in[12];
    const float* p_g    = (const float*)d_in[13];
    const float* p_b    = (const float*)d_in[14];
    const float* p_m    = (const float*)d_in[15];
    const float* p_v    = (const float*)d_in[16];

    float* y_out    = (float*)d_out;                                 // [T,B,C,N]
    float* attn_out = y_out + (size_t)TT * BB * CC * NN;             // [T,B,H,N]

    // ---- workspace carve-up ----
    char* ws = (char*)d_ws;
    size_t off = 0;
    const size_t big = (size_t)TT * BB * NN * CC * sizeof(_Float16); // 64 MiB each
    _Float16* xsT = (_Float16*)(ws + off); off += big;
    _Float16* qT  = (_Float16*)(ws + off); off += big;
    _Float16* kT  = (_Float16*)(ws + off); off += big;
    _Float16* yT  = (_Float16*)(ws + off); off += big;
    _Float16* wqh = (_Float16*)(ws + off); off += (size_t)CC * CC * sizeof(_Float16);
    _Float16* wkh = (_Float16*)(ws + off); off += (size_t)CC * CC * sizeof(_Float16);
    _Float16* wph = (_Float16*)(ws + off); off += (size_t)CC * CC * sizeof(_Float16);
    float* qscale = (float*)(ws + off); off += CC * sizeof(float);
    float* qbias  = (float*)(ws + off); off += CC * sizeof(float);
    float* kscale = (float*)(ws + off); off += CC * sizeof(float);
    float* kbias  = (float*)(ws + off); off += CC * sizeof(float);
    float* pscale = (float*)(ws + off); off += CC * sizeof(float);
    float* pbias  = (float*)(ws + off); off += CC * sizeof(float);

    // 1) weight conversion + BN folding
    prep_kernel<<<(CC * CC + 255) / 256, 256, 0, stream>>>(
        q_w, k_w, proj_w, q_g, q_b, q_m, q_v, k_g, k_b, k_m, k_v,
        proj_b, p_g, p_b, p_m, p_v,
        wqh, wkh, wph, qscale, qbias, kscale, kbias, pscale, pbias);

    // 2) LIF on x + transpose to [T,B,N,C]
    lif_transpose_kernel<<<dim3(NN / 32, CC / 32, BB), dim3(32, 8), 0, stream>>>(x, xsT);

    // 3) fused q/k GEMM + BN + LIF -> spike matrices
    qk_gemm_lif_kernel<<<dim3(NN / 32, CC / 128, BB), 256, 0, stream>>>(
        xsT, wqh, wkh, qscale, qbias, kscale, kbias, qT, kT);

    // 4) head-sum + attn LIF + y = attn * k
    attn_y_kernel<<<(BB * HEADS * NN + 255) / 256, 256, 0, stream>>>(qT, kT, yT, attn_out);

    // 5) projection GEMM + bias + BN -> fp32 output
    proj_gemm_kernel<<<dim3(NN / 32, CC / 128, TT * BB), 256, 0, stream>>>(
        yT, wph, pscale, pbias, y_out);
}
